// MultiHeadAttention_58480274702375
// MI455X (gfx1250) — matrix operand order repaired and checked by reading, not compiled, not run on hardware
//
#include <hip/hip_runtime.h>
#include <hip/hip_bf16.h>

#define BATCH  2
#define LSEQ   2048
#define DMODEL 512
#define NHEAD  8
#define DHEAD  64

typedef __bf16 bf16_t;
typedef bf16_t v16bf __attribute__((ext_vector_type(16)));
typedef bf16_t v8bf  __attribute__((ext_vector_type(8)));
typedef bf16_t v4bf  __attribute__((ext_vector_type(4)));
typedef float  v8f   __attribute__((ext_vector_type(8)));

union Frag16 { v16bf v; v8bf h[2]; };

__device__ __forceinline__ v8f wmma_bf16(const v16bf& a, const v16bf& b, const v8f& c) {
  // D = A(16x32 bf16) x B(32x16 bf16) + C(16x16 f32)
  return __builtin_amdgcn_wmma_f32_16x16x32_bf16(false, a, false, b, (short)0, c, false, false);
}

// LDS byte address of a __shared__ pointer (flat LDS addr truncates to [31:0])
__device__ __forceinline__ unsigned lds_addr(const void* p) {
  return (unsigned)(unsigned long long)p;
}

// CDNA5 async copy: 16B per lane, global -> LDS, tracked by ASYNCcnt (GVS mode)
__device__ __forceinline__ void async_ld_b128(unsigned lds, unsigned voff, const void* sbase) {
  asm volatile("global_load_async_to_lds_b128 %0, %1, %2"
               :: "v"(lds), "v"(voff), "s"((unsigned long long)sbase)
               : "memory");
}

// ---------------------------------------------------------------------------
// GEMM: acc[M,N] = A[M,K] @ B[K,N]   (fp32 in, bf16 WMMA, f32 accumulate)
// MODE 0: C[m][n]  = acc                                  (fp32, Q projection)
// MODE 1: C[m][n]  = resid + relu(acc + bias[n])          (fused FFN tail)
// MODE 2: Cb[m][n] = bf16(acc)                            (K projection)
// MODE 3: Cb[b][h][d][key] = bf16(acc)  (transposed)      (V projection)
// Block tile 128x64, 8 waves, each wave 32x32 (2x2 WMMA tiles), K-step 32.
// ---------------------------------------------------------------------------
template <int MODE>
__global__ __launch_bounds__(256)
void gemm_kernel(const float* __restrict__ A, const float* __restrict__ Bm,
                 const float* __restrict__ resid, const float* __restrict__ bias,
                 float* __restrict__ C, bf16_t* __restrict__ Cb, int M, int N, int K)
{
  constexpr int TM = 128, TN = 64, TK = 32, LDA = TK + 8, LDB = TK + 8;
  __shared__ __align__(16) bf16_t As[TM][LDA];   // [m][k]
  __shared__ __align__(16) bf16_t Bs[TN][LDB];   // transposed: [n][k]

  const int tid  = threadIdx.x;
  const int lane = tid & 31, w = tid >> 5;
  const int wm = w & 3, wn = w >> 2;             // wave grid 4(M) x 2(N)
  const int half = lane >> 4, l16 = lane & 15;
  const int m0 = blockIdx.y * TM, n0 = blockIdx.x * TN;

  v8f acc[2][2] = {};

  for (int k0 = 0; k0 < K; k0 += TK) {
    // ---- stage A tile 128x32 (fp32 -> bf16), float4 coalesced
#pragma unroll
    for (int it = 0; it < 4; ++it) {
      int c   = it * 256 + tid;        // 0..1023 float4 chunks
      int row = c >> 3;                // 8 float4 per 32-float row
      int col = (c & 7) << 2;
      const float4 f = *(const float4*)(A + (size_t)(m0 + row) * K + k0 + col);
      v4bf pk; pk.x = (bf16_t)f.x; pk.y = (bf16_t)f.y; pk.z = (bf16_t)f.z; pk.w = (bf16_t)f.w;
      *(v4bf*)&As[row][col] = pk;
    }
    // ---- stage B tile 32x64, transposed into [n][k]
#pragma unroll
    for (int it = 0; it < 2; ++it) {
      int c  = it * 256 + tid;         // 0..511 float4 chunks
      int kr = c >> 4;                 // 16 float4 per 64-float row
      int nc = (c & 15) << 2;
      const float4 f = *(const float4*)(Bm + (size_t)(k0 + kr) * N + n0 + nc);
      Bs[nc + 0][kr] = (bf16_t)f.x;
      Bs[nc + 1][kr] = (bf16_t)f.y;
      Bs[nc + 2][kr] = (bf16_t)f.z;
      Bs[nc + 3][kr] = (bf16_t)f.w;
    }
    __syncthreads();

    // ---- fragments (b128 LDS loads; padded rows keep 16B alignment)
    Frag16 a[2], b[2];
#pragma unroll
    for (int t = 0; t < 2; ++t) {
      const bf16_t* ap = &As[wm * 32 + t * 16 + l16][0];
      a[t].h[0] = *(const v8bf*)(ap + 8 * half);        // K = 8*half + 0..7
      a[t].h[1] = *(const v8bf*)(ap + 8 * half + 16);   // K = 8*half + 16..23
      const bf16_t* bp = &Bs[wn * 32 + t * 16 + l16][0];
      b[t].h[0] = *(const v8bf*)(bp + 8 * half);        // K = 8*half + 0..7
      b[t].h[1] = *(const v8bf*)(bp + 8 * half + 16);   // K = 8*half + 16..23
    }
#pragma unroll
    for (int i = 0; i < 2; ++i)
#pragma unroll
      for (int j = 0; j < 2; ++j)
        acc[i][j] = wmma_bf16(a[i].v, b[j].v, acc[i][j]);
    __syncthreads();
  }

  // ---- epilogue (C layout: lane<16 -> M=r, N=lane; lane>=16 -> M=r+8)
#pragma unroll
  for (int i = 0; i < 2; ++i) {
#pragma unroll
    for (int j = 0; j < 2; ++j) {
      int col = n0 + wn * 32 + j * 16 + l16;
      int rb  = m0 + wm * 32 + i * 16 + half * 8;
      if (MODE == 3) {
        // transposed bf16 store: rows rb..rb+7 are 8 consecutive keys -> 16B
        int bb_ = rb >> 11;            // / LSEQ
        int key = rb & (LSEQ - 1);
        int hh  = col >> 6;            // / DHEAD
        int dd  = col & (DHEAD - 1);
        v8bf pk;
#pragma unroll
        for (int r = 0; r < 8; ++r) pk[r] = (bf16_t)acc[i][j][r];
        *(v8bf*)(Cb + ((size_t)((bb_ * NHEAD + hh) * DHEAD + dd)) * LSEQ + key) = pk;
      } else {
#pragma unroll
        for (int r = 0; r < 8; ++r) {
          size_t idx = (size_t)(rb + r) * N + col;
          float v = acc[i][j][r];
          if (MODE == 0) {
            C[idx] = v;
          } else if (MODE == 1) {
            float z = v + bias[col];
            C[idx] = resid[idx] + fmaxf(z, 0.f);
          } else {            // MODE 2: bf16 row-major
            Cb[idx] = (bf16_t)v;
          }
        }
      }
    }
  }
}

// ---------------------------------------------------------------------------
// Flash attention: per (b, h, 128 q-rows) workgroup; 8 waves x 16 q-rows.
// K (bf16 [b][key][h*64+d]) and V (bf16 [b][h][d][key], pre-transposed) are
// staged into LDS with GLOBAL_LOAD_ASYNC_TO_LDS_B128 (ASYNCcnt).
// Writes tmp = qf + softmax(Q Kt / sqrt(D)) @ V   (residual fused)
// ---------------------------------------------------------------------------
__global__ __launch_bounds__(256)
void flash_kernel(const float* __restrict__ qf, const bf16_t* __restrict__ kbf,
                  const bf16_t* __restrict__ vt, float* __restrict__ tmp)
{
  constexpr int LDK = DHEAD + 8;                      // 72 elems = 144B rows (16B mult.)
  __shared__ __align__(16) bf16_t Kt[64][LDK];        // [key][d]
  __shared__ __align__(16) bf16_t Vt[DHEAD][LDK];     // [d][key]
  __shared__ __align__(16) bf16_t Ps[8][16][LDK];     // per-wave P staging

  const int tid = threadIdx.x, lane = tid & 31, w = tid >> 5;
  const int half = lane >> 4, l16 = lane & 15;
  const int h = blockIdx.y, b = blockIdx.z;
  const int q0 = blockIdx.x * 128 + w * 16;

  const float scale = 0.04419417382415922f;           // 1/sqrt(512)

  // Q fragments (A layout 16x32), pre-scaled by 1/sqrt(D); two d-chunks of 32
  Frag16 qa[2];
  {
    const float* qp = qf + (size_t)(b * LSEQ + q0 + l16) * DMODEL + h * DHEAD;
#pragma unroll
    for (int ds = 0; ds < 2; ++ds) {
#pragma unroll
      for (int e = 0; e < 8; ++e) {
        qa[ds].v[e]     = (bf16_t)(qp[ds * 32 + 8 * half + e] * scale);
        qa[ds].v[e + 8] = (bf16_t)(qp[ds * 32 + 8 * half + 16 + e] * scale);
      }
    }
  }

  v8f O[4] = {};
  float m_[8], l_[8];
#pragma unroll
  for (int r = 0; r < 8; ++r) { m_[r] = -1e30f; l_[r] = 0.f; }

  for (int kb = 0; kb < LSEQ; kb += 64) {
    // ---- async stage K/V tiles (pure bf16 byte-moves, 16B per lane)
    const bf16_t* kbase = kbf + (size_t)(b * LSEQ + kb) * DMODEL + h * DHEAD;
    const bf16_t* vbase = vt + ((size_t)(b * NHEAD + h) * DHEAD) * LSEQ + kb;
#pragma unroll
    for (int it = 0; it < 2; ++it) {
      int c   = it * 256 + tid;                       // 0..511 16B chunks
      int row = c >> 3;                               // key (K) / d (V)
      int c16 = c & 7;
      async_ld_b128(lds_addr(&Kt[row][c16 * 8]),
                    (unsigned)(row * (DMODEL * 2) + c16 * 16), kbase);
      async_ld_b128(lds_addr(&Vt[row][c16 * 8]),
                    (unsigned)(row * (LSEQ * 2) + c16 * 16), vbase);
    }
    asm volatile("s_wait_asynccnt 0" ::: "memory");
    __syncthreads();

    // ---- S = (Q*scale) @ K^T : 4 key subtiles of 16, K-dim = d (2x32)
    v8f s[4];
#pragma unroll
    for (int kt = 0; kt < 4; ++kt) {
      const bf16_t* kp = &Kt[kt * 16 + l16][0];
      Frag16 k0, k1;
      k0.h[0] = *(const v8bf*)(kp + 8 * half);
      k0.h[1] = *(const v8bf*)(kp + 8 * half + 16);
      k1.h[0] = *(const v8bf*)(kp + 32 + 8 * half);
      k1.h[1] = *(const v8bf*)(kp + 32 + 8 * half + 16);
      v8f t = {};
      t = wmma_bf16(qa[0].v, k0.v, t);
      t = wmma_bf16(qa[1].v, k1.v, t);
      s[kt] = t;
    }

    // ---- online softmax (row reductions across 16-lane halves)
#pragma unroll
    for (int r = 0; r < 8; ++r) {
      float mx = fmaxf(fmaxf(s[0][r], s[1][r]), fmaxf(s[2][r], s[3][r]));
#pragma unroll
      for (int msk = 1; msk < 16; msk <<= 1)
        mx = fmaxf(mx, __shfl_xor(mx, msk, 32));
      float mn = fmaxf(m_[r], mx);
      float alpha = __expf(m_[r] - mn);
      float sum = 0.f;
#pragma unroll
      for (int kt = 0; kt < 4; ++kt) {
        float p = __expf(s[kt][r] - mn);
        s[kt][r] = p;
        sum += p;
      }
#pragma unroll
      for (int msk = 1; msk < 16; msk <<= 1)
        sum += __shfl_xor(sum, msk, 32);
      l_[r] = l_[r] * alpha + sum;
      m_[r] = mn;
#pragma unroll
      for (int dt = 0; dt < 4; ++dt) O[dt][r] *= alpha;
    }

    // ---- C-layout -> A-layout re-staging of P through per-wave LDS
#pragma unroll
    for (int kt = 0; kt < 4; ++kt)
#pragma unroll
      for (int r = 0; r < 8; ++r)
        Ps[w][half * 8 + r][kt * 16 + l16] = (bf16_t)s[kt][r];
    asm volatile("s_wait_dscnt 0" ::: "memory");   // same-wave LDS RAW fence

    Frag16 pa[2];
#pragma unroll
    for (int kc = 0; kc < 2; ++kc) {
      const bf16_t* pp = &Ps[w][l16][kc * 32];
      pa[kc].h[0] = *(const v8bf*)(pp + 8 * half);
      pa[kc].h[1] = *(const v8bf*)(pp + 8 * half + 16);
    }

    // ---- O += P @ V   (4 d-subtiles of 16, K-dim = keys 2x32)
#pragma unroll
    for (int dt = 0; dt < 4; ++dt) {
      const bf16_t* vp = &Vt[dt * 16 + l16][0];
      Frag16 v0, v1;
      v0.h[0] = *(const v8bf*)(vp + 8 * half);
      v0.h[1] = *(const v8bf*)(vp + 8 * half + 16);
      v1.h[0] = *(const v8bf*)(vp + 32 + 8 * half);
      v1.h[1] = *(const v8bf*)(vp + 32 + 8 * half + 16);
      O[dt] = wmma_bf16(pa[0].v, v0.v, O[dt]);
      O[dt] = wmma_bf16(pa[1].v, v1.v, O[dt]);
    }
    __syncthreads();
  }

  // ---- normalize, fuse residual tmp = qf + O/l
#pragma unroll
  for (int dt = 0; dt < 4; ++dt) {
    int d = dt * 16 + l16;
#pragma unroll
    for (int r = 0; r < 8; ++r) {
      int q = q0 + half * 8 + r;
      size_t idx = (size_t)(b * LSEQ + q) * DMODEL + h * DHEAD + d;
      tmp[idx] = qf[idx] + O[dt][r] / l_[r];
    }
  }
}

// ---------------------------------------------------------------------------
extern "C" void kernel_launch(void* const* d_in, const int* in_sizes, int n_in,
                              void* d_out, int out_size, void* d_ws, size_t ws_size,
                              hipStream_t stream) {
  (void)in_sizes; (void)n_in; (void)out_size; (void)ws_size;
  const float* query = (const float*)d_in[0];
  const float* key   = (const float*)d_in[1];
  const float* value = (const float*)d_in[2];
  const float* Wq    = (const float*)d_in[3];
  const float* Wk    = (const float*)d_in[4];
  const float* Wv    = (const float*)d_in[5];
  const float* Wo    = (const float*)d_in[6];
  const float* bo    = (const float*)d_in[7];
  float* out = (float*)d_out;

  const size_t MD = (size_t)BATCH * LSEQ * DMODEL;   // 4096*512 elements
  float*  qf  = (float*)d_ws;                        // projected Q (fp32, residual)
  float*  tmp = qf + MD;                             // q + attention(o)
  bf16_t* kbf = (bf16_t*)(tmp + MD);                 // projected K (bf16 [m][n])
  bf16_t* vt  = kbf + MD;                            // projected V (bf16 [b][h][d][key])

  const int M = BATCH * LSEQ;                        // 4096
  dim3 gg(DMODEL / 64, M / 128);                     // (8, 32)
  dim3 bb(256);
  gemm_kernel<0><<<gg, bb, 0, stream>>>(query, Wq, nullptr, nullptr, qf,  nullptr, M, DMODEL, DMODEL);
  gemm_kernel<2><<<gg, bb, 0, stream>>>(key,   Wk, nullptr, nullptr, nullptr, kbf, M, DMODEL, DMODEL);
  gemm_kernel<3><<<gg, bb, 0, stream>>>(value, Wv, nullptr, nullptr, nullptr, vt,  M, DMODEL, DMODEL);

  dim3 ga(LSEQ / 128, NHEAD, BATCH);                 // (16, 8, 2)
  flash_kernel<<<ga, bb, 0, stream>>>(qf, kbf, vt, tmp);

  gemm_kernel<1><<<gg, bb, 0, stream>>>(tmp, Wo, tmp, bo, out, nullptr, M, DMODEL, DMODEL);
}
